// CausalSelfAttention_11192684774089
// MI455X (gfx1250) — matrix operand order repaired and checked by reading, not compiled, not run on hardware
//
#include <hip/hip_runtime.h>
#include <hip/hip_bf16.h>

// ---------------------------------------------------------------------------
// Types for WMMA fragments (gfx1250, wave32)
// ---------------------------------------------------------------------------
typedef __attribute__((ext_vector_type(16))) __bf16        v16bf;
typedef __attribute__((ext_vector_type(8)))  float         f32x8;
typedef __attribute__((ext_vector_type(4)))  float         f32x4;
typedef __attribute__((ext_vector_type(4)))  unsigned int  u32x4;
typedef __attribute__((ext_vector_type(4)))  int           i32x4;
typedef __attribute__((ext_vector_type(2)))  unsigned int  u32x2;

union Frag16 {           // 16 bf16 elements = one A or B operand (8 VGPRs)
    v16bf bf;
    u32x4 u[2];
};

__device__ __forceinline__ f32x8 zero8() {
    f32x8 z;
#pragma unroll
    for (int i = 0; i < 8; i++) z[i] = 0.0f;
    return z;
}

__device__ __forceinline__ float bf2f(unsigned short u) {
    return __uint_as_float(((unsigned int)u) << 16);
}
__device__ __forceinline__ unsigned short f2bf(float f) {
    unsigned int u = __float_as_uint(f);
    u += 0x7FFFu + ((u >> 16) & 1u);     // round-to-nearest-even
    return (unsigned short)(u >> 16);
}

__device__ __forceinline__ f32x8 wmma_bf16(const v16bf& a, const v16bf& b, const f32x8& c) {
    return __builtin_amdgcn_wmma_f32_16x16x32_bf16(
        /*neg_a=*/false, a, /*neg_b=*/false, b,
        /*c_mod=*/(short)0, c, /*reuse_a=*/false, /*reuse_b=*/false);
}

// ---------------------------------------------------------------------------
// CDNA5 async global->LDS path (ASYNCcnt), with synchronous fallback
// ---------------------------------------------------------------------------
#if defined(__has_builtin)
#if __has_builtin(__builtin_amdgcn_global_load_async_to_lds_b128) && \
    __has_builtin(__builtin_amdgcn_s_wait_asynccnt)
#define USE_ASYNC_LDS 1
#endif
#endif
#ifndef USE_ASYNC_LDS
#define USE_ASYNC_LDS 0
#endif

typedef __attribute__((address_space(1))) i32x4* gbl_i4ptr;
typedef __attribute__((address_space(3))) i32x4* lds_i4ptr;

__device__ __forceinline__ void copy16_g2l(const unsigned short* g, unsigned short* l) {
#if USE_ASYNC_LDS
    __builtin_amdgcn_global_load_async_to_lds_b128(
        (gbl_i4ptr)(i32x4*)(g), (lds_i4ptr)(i32x4*)(l), /*offset=*/0, /*cpol=*/0);
#else
    *(u32x4*)l = *(const u32x4*)g;
#endif
}
__device__ __forceinline__ void wait_async_all() {
#if USE_ASYNC_LDS
    __builtin_amdgcn_s_wait_asynccnt(0);
#endif
}

// ---------------------------------------------------------------------------
// Problem constants
// ---------------------------------------------------------------------------
#define NB     4
#define NT     2048
#define NC     2048
#define NH     16
#define ND     128
#define N3C    6144
#define LORA_R 8
#define LORA_SCALE 0.125f
#define ATT_SCALE  0.08838834764831845f   // 1/sqrt(128)

// ---------------------------------------------------------------------------
// Elementwise f32 -> bf16 conversion (8 elems / thread)
// ---------------------------------------------------------------------------
__global__ __launch_bounds__(256) void convert_f32_to_bf16_kernel(
    const float* __restrict__ src, unsigned short* __restrict__ dst, size_t n)
{
    size_t i = ((size_t)blockIdx.x * 256 + threadIdx.x) * 8;
    if (i + 8 <= n) {
        f32x4 a = *(const f32x4*)&src[i];
        f32x4 b = *(const f32x4*)&src[i + 4];
        u32x4 o;
        o.x = (unsigned)f2bf(a.x) | ((unsigned)f2bf(a.y) << 16);
        o.y = (unsigned)f2bf(a.z) | ((unsigned)f2bf(a.w) << 16);
        o.z = (unsigned)f2bf(b.x) | ((unsigned)f2bf(b.y) << 16);
        o.w = (unsigned)f2bf(b.z) | ((unsigned)f2bf(b.w) << 16);
        *(u32x4*)&dst[i] = o;
    }
}

// ---------------------------------------------------------------------------
// Transpose + convert: dst[C][R] (bf16) = src[R][C] (f32)
// ---------------------------------------------------------------------------
__global__ __launch_bounds__(256) void transpose_f32_to_bf16(
    const float* __restrict__ src, unsigned short* __restrict__ dst, int R, int C)
{
    __shared__ float tile[32][33];
    int tx = threadIdx.x & 31, ty = threadIdx.x >> 5;
    int r0 = blockIdx.y * 32, c0 = blockIdx.x * 32;
#pragma unroll
    for (int i = ty; i < 32; i += 8)
        tile[i][tx] = src[(size_t)(r0 + i) * C + (c0 + tx)];
    __syncthreads();
#pragma unroll
    for (int i = ty; i < 32; i += 8)
        dst[(size_t)(c0 + i) * R + (r0 + tx)] = f2bf(tile[tx][i]);
}

// bf16 -> bf16 transpose (per batch plane): dst[C][R] = src[R][C]
__global__ __launch_bounds__(256) void transpose_bf16(
    const unsigned short* __restrict__ src, unsigned short* __restrict__ dst, int R, int C)
{
    __shared__ unsigned short tile[32][33];
    size_t off = (size_t)blockIdx.z * R * C;
    int tx = threadIdx.x & 31, ty = threadIdx.x >> 5;
    int r0 = blockIdx.y * 32, c0 = blockIdx.x * 32;
#pragma unroll
    for (int i = ty; i < 32; i += 8)
        tile[i][tx] = src[off + (size_t)(r0 + i) * C + (c0 + tx)];
    __syncthreads();
#pragma unroll
    for (int i = ty; i < 32; i += 8)
        dst[off + (size_t)(c0 + i) * R + (r0 + tx)] = tile[tx][i];
}

// ---------------------------------------------------------------------------
// WMMA GEMM:  C[M][N] = A[M][K](bf16) * BT[N][K]^T(bf16) + bias[N]
//   block tile 128x256, 8 waves (2x4), wave tile 64x64, K-step 32,
//   double-buffered LDS fed by CDNA5 async global->LDS copies.
// ---------------------------------------------------------------------------
template <bool C_BF16>
__global__ __launch_bounds__(256) void gemm_wmma_kernel(
    const unsigned short* __restrict__ A,
    const unsigned short* __restrict__ BT,
    const float* __restrict__ bias,
    void* __restrict__ Cptr,
    int M, int N, int K)
{
    __shared__ __align__(16) unsigned short As[2][128][40];
    __shared__ __align__(16) unsigned short Bs[2][256][40];

    const int tid  = threadIdx.x;
    const int lane = tid & 31;
    const int wave = tid >> 5;
    const int lm = lane & 15, lh = lane >> 4;
    const int wm = wave >> 2;          // 0..1 : 64-row group
    const int wn = wave & 3;           // 0..3 : 64-col group
    const int m0 = blockIdx.y * 128;
    const int n0 = blockIdx.x * 256;

    // cooperative tile fetch: A = 512 16B chunks, B = 1024 16B chunks
    auto load_tiles = [&](int buf, int k0) {
#pragma unroll
        for (int j = 0; j < 2; j++) {
            int c = tid + j * 256;            // A: 128 rows x 32 k
            int row = c >> 2;
            int kc  = (c & 3) * 8;
            copy16_g2l(&A[(size_t)(m0 + row) * K + k0 + kc], &As[buf][row][kc]);
        }
#pragma unroll
        for (int j = 0; j < 4; j++) {
            int c = tid + j * 256;            // B: 256 N-rows x 32 k
            int row = c >> 2;
            int kc  = (c & 3) * 8;
            copy16_g2l(&BT[(size_t)(n0 + row) * K + k0 + kc], &Bs[buf][row][kc]);
        }
    };

    f32x8 acc[4][4];
#pragma unroll
    for (int mi = 0; mi < 4; mi++)
#pragma unroll
        for (int ni = 0; ni < 4; ni++) acc[mi][ni] = zero8();

    load_tiles(0, 0);
    wait_async_all();
    __syncthreads();

    const int KSTEPS = K >> 5;
    for (int ks = 0; ks < KSTEPS; ks++) {
        const int buf = ks & 1;
        if (ks + 1 < KSTEPS)
            load_tiles(buf ^ 1, (ks + 1) * 32);   // async DMA overlaps compute

        Frag16 af[4];
#pragma unroll
        for (int mi = 0; mi < 4; mi++) {
            int row = wm * 64 + mi * 16 + lm;
            af[mi].u[0] = *(const u32x4*)&As[buf][row][lh * 8];
            af[mi].u[1] = *(const u32x4*)&As[buf][row][lh * 8 + 16];
        }
        Frag16 bf[4];
#pragma unroll
        for (int ni = 0; ni < 4; ni++) {
            int row = wn * 64 + ni * 16 + lm;
            bf[ni].u[0] = *(const u32x4*)&Bs[buf][row][lh * 8];
            bf[ni].u[1] = *(const u32x4*)&Bs[buf][row][lh * 8 + 16];
        }
#pragma unroll
        for (int mi = 0; mi < 4; mi++)
#pragma unroll
            for (int ni = 0; ni < 4; ni++)
                acc[mi][ni] = wmma_bf16(af[mi].bf, bf[ni].bf, acc[mi][ni]);

        if (ks + 1 < KSTEPS) {
            wait_async_all();
            __syncthreads();
        }
    }

    // ---- epilogue: C frag: lane col = lm, rows = lh*8 + r ----
#pragma unroll
    for (int mi = 0; mi < 4; mi++) {
#pragma unroll
        for (int ni = 0; ni < 4; ni++) {
            int col = n0 + wn * 64 + ni * 16 + lm;
            float bv = bias ? bias[col] : 0.0f;
#pragma unroll
            for (int r = 0; r < 8; r++) {
                int row = m0 + wm * 64 + mi * 16 + lh * 8 + r;
                float v = acc[mi][ni][r] + bv;
                if (C_BF16)
                    ((unsigned short*)Cptr)[(size_t)row * N + col] = f2bf(v);
                else
                    ((float*)Cptr)[(size_t)row * N + col] = v;
            }
        }
    }
}

// ---------------------------------------------------------------------------
// LoRA rank-8 reductions: tq[row][r] = sum_c q[row][c]*Aq[c][r]  (same for k)
// ---------------------------------------------------------------------------
__global__ __launch_bounds__(256) void lora_reduce_kernel(
    const unsigned short* __restrict__ qkv,
    const float* __restrict__ Aq, const float* __restrict__ Ak,
    float* __restrict__ tq, float* __restrict__ tk)
{
    int row = blockIdx.x;
    const unsigned short* qr = qkv + (size_t)row * N3C;
    float aq[LORA_R], ak[LORA_R];
#pragma unroll
    for (int r = 0; r < LORA_R; r++) { aq[r] = 0.f; ak[r] = 0.f; }
    for (int c = threadIdx.x; c < NC; c += 256) {
        float qv = bf2f(qr[c]);
        float kv = bf2f(qr[NC + c]);
#pragma unroll
        for (int r = 0; r < LORA_R; r++) {
            aq[r] += qv * Aq[(size_t)c * LORA_R + r];
            ak[r] += kv * Ak[(size_t)c * LORA_R + r];
        }
    }
    int lane = threadIdx.x & 31, warp = threadIdx.x >> 5;
#pragma unroll
    for (int r = 0; r < LORA_R; r++) {
#pragma unroll
        for (int off = 16; off > 0; off >>= 1) {
            aq[r] += __shfl_xor(aq[r], off, 32);
            ak[r] += __shfl_xor(ak[r], off, 32);
        }
    }
    __shared__ float red[8][16];
    if (lane == 0) {
#pragma unroll
        for (int r = 0; r < LORA_R; r++) {
            red[warp][r]          = aq[r];
            red[warp][LORA_R + r] = ak[r];
        }
    }
    __syncthreads();
    if (threadIdx.x < 16) {
        float s = 0.f;
#pragma unroll
        for (int w = 0; w < 8; w++) s += red[w][threadIdx.x];
        if (threadIdx.x < LORA_R) tq[(size_t)row * LORA_R + threadIdx.x] = s;
        else                      tk[(size_t)row * LORA_R + threadIdx.x - LORA_R] = s;
    }
}

// ---------------------------------------------------------------------------
// Fixup: add LoRA-B, apply RoPE, reshape to head-major bf16 (qh/kh/vh)
// ---------------------------------------------------------------------------
__global__ __launch_bounds__(128) void fixup_kernel(
    const unsigned short* __restrict__ qkv,
    const float* __restrict__ tq, const float* __restrict__ tk,
    const float* __restrict__ Bq, const float* __restrict__ Bk,
    unsigned short* __restrict__ qh, unsigned short* __restrict__ kh,
    unsigned short* __restrict__ vh)
{
    int row = blockIdx.x;
    int b = row / NT, t = row % NT;
    int d = threadIdx.x;                    // 0..127
    __shared__ float sq[ND], sk[ND];

    float tqv[LORA_R], tkv[LORA_R];
#pragma unroll
    for (int r = 0; r < LORA_R; r++) {
        tqv[r] = tq[(size_t)row * LORA_R + r];
        tkv[r] = tk[(size_t)row * LORA_R + r];
    }
    float inv_freq = __expf(-9.210340371976184f * (float)(d & 63) / 64.0f);
    float ang = (float)t * inv_freq;
    float cs = __cosf(ang), sn = __sinf(ang);

    const unsigned short* qr = qkv + (size_t)row * N3C;
    for (int h = 0; h < NH; h++) {
        int col = h * ND + d;
        float qv = bf2f(qr[col]);
        float kv = bf2f(qr[NC + col]);
        float vv = bf2f(qr[2 * NC + col]);
        float lq = 0.f, lk = 0.f;
#pragma unroll
        for (int r = 0; r < LORA_R; r++) {
            lq += tqv[r] * Bq[(size_t)r * NC + col];
            lk += tkv[r] * Bk[(size_t)r * NC + col];
        }
        qv += LORA_SCALE * lq;
        kv += LORA_SCALE * lk;
        sq[d] = qv; sk[d] = kv;
        __syncthreads();
        float rq = (d < 64) ? -sq[2 * d + 1] : sq[2 * (d - 64)];
        float rk = (d < 64) ? -sk[2 * d + 1] : sk[2 * (d - 64)];
        float qo = qv * cs + rq * sn;
        float ko = kv * cs + rk * sn;
        size_t hidx = (((size_t)b * NH + h) * NT + t) * ND + d;
        qh[hidx] = f2bf(qo);
        kh[hidx] = f2bf(ko);
        vh[hidx] = f2bf(vv);
        __syncthreads();
    }
}

// ---------------------------------------------------------------------------
// Flash attention (causal, online softmax), WMMA bf16, f32 accum.
//   grid = B*H*(T/128) blocks, 256 threads (8 waves, 16 q-rows per wave).
//   K/V fragments come straight from global: 96 MB working set fits the
//   192 MB L2, so no LDS staging needed; prefetch warms the next K block.
// ---------------------------------------------------------------------------
__global__ __launch_bounds__(256) void flash_attn_kernel(
    const unsigned short* __restrict__ qh,
    const unsigned short* __restrict__ kh,
    const unsigned short* __restrict__ vt,   // [b][h][d][t]
    unsigned short* __restrict__ y)          // [b][t][h*128+d]
{
    int bid  = blockIdx.x;
    int qblk = bid & 15;          // T/128 = 16
    int bh   = bid >> 4;          // 0..63
    int b = bh >> 4, h = bh & 15;
    int wave = threadIdx.x >> 5, lane = threadIdx.x & 31;
    int lm = lane & 15, lh = lane >> 4;
    int qbase = qblk * 128 + wave * 16;

    const size_t plane = (size_t)bh * NT * ND;
    const unsigned short* Q  = qh + plane;
    const unsigned short* Kp = kh + plane;
    const unsigned short* Vt = vt + plane;

    // Q fragments: 16 rows x 128 d = 4 A-frags, resident for the whole loop
    Frag16 qf[4];
    {
        int qrow = qbase + lm;
#pragma unroll
        for (int kc = 0; kc < 4; kc++) {
            const unsigned short* p = Q + (size_t)qrow * ND + kc * 32 + lh * 8;
            qf[kc].u[0] = *(const u32x4*)p;
            qf[kc].u[1] = *(const u32x4*)(p + 16);
        }
    }

    f32x8 o[8];
#pragma unroll
    for (int dt = 0; dt < 8; dt++) o[dt] = zero8();
    float m[8], l[8];
#pragma unroll
    for (int r = 0; r < 8; r++) { m[r] = -1e30f; l[r] = 0.f; }

    __shared__ __align__(16) unsigned short Pb[8][16][72];

    int kbmax = (qbase + 15) >> 6;
    for (int kb = 0; kb <= kbmax; kb++) {
        // warm L2/WGP$ for the next K block (global_prefetch_b8)
        if (kb < kbmax) {
            const unsigned short* nk = Kp + (size_t)((kb + 1) * 64 + lane * 2) * ND;
            __builtin_prefetch(nk, 0, 1);
            __builtin_prefetch(nk + ND, 0, 1);
        }

        // ---- S = Q * K^T over this 64-col block ----
        f32x8 s[4];
#pragma unroll
        for (int nt = 0; nt < 4; nt++) s[nt] = zero8();
#pragma unroll
        for (int nt = 0; nt < 4; nt++) {
            int kcol = kb * 64 + nt * 16 + lm;          // key time index
#pragma unroll
            for (int kc = 0; kc < 4; kc++) {
                Frag16 bf;
                const unsigned short* p = Kp + (size_t)kcol * ND + kc * 32 + lh * 8;
                bf.u[0] = *(const u32x4*)p;
                bf.u[1] = *(const u32x4*)(p + 16);
                s[nt] = wmma_bf16(qf[kc].bf, bf.bf, s[nt]);
            }
        }

        // ---- scale, causal mask, online softmax stats ----
        float mnew[8];
#pragma unroll
        for (int r = 0; r < 8; r++) {
            int R = qbase + lh * 8 + r;
            float mx = m[r];
#pragma unroll
            for (int nt = 0; nt < 4; nt++) {
                int Cg = kb * 64 + nt * 16 + lm;
                float v = s[nt][r] * ATT_SCALE;
                if (Cg > R) v = -1e30f;
                s[nt][r] = v;
                mx = fmaxf(mx, v);
            }
            mnew[r] = mx;
        }
#pragma unroll
        for (int r = 0; r < 8; r++) {
#pragma unroll
            for (int off = 1; off < 16; off <<= 1)
                mnew[r] = fmaxf(mnew[r], __shfl_xor(mnew[r], off, 32));
        }
#pragma unroll
        for (int r = 0; r < 8; r++) {
            float alpha = __expf(m[r] - mnew[r]);
            m[r] = mnew[r];
            float rs = 0.f;
#pragma unroll
            for (int nt = 0; nt < 4; nt++) {
                float p = __expf(s[nt][r] - m[r]);
                s[nt][r] = p;
                rs += p;
            }
#pragma unroll
            for (int off = 1; off < 16; off <<= 1)
                rs += __shfl_xor(rs, off, 32);
            l[r] = l[r] * alpha + rs;
#pragma unroll
            for (int dt = 0; dt < 8; dt++) o[dt][r] *= alpha;
            // stash P (C-frag layout -> LDS) for re-load as A-frag
#pragma unroll
            for (int nt = 0; nt < 4; nt++)
                Pb[wave][lh * 8 + r][nt * 16 + lm] = f2bf(s[nt][r]);
        }

        // ---- O += P * V ----
        Frag16 pf[2];
#pragma unroll
        for (int kc2 = 0; kc2 < 2; kc2++) {
            int k0 = kc2 * 32 + lh * 8;
            pf[kc2].u[0] = *(const u32x4*)&Pb[wave][lm][k0];
            pf[kc2].u[1] = *(const u32x4*)&Pb[wave][lm][k0 + 16];
        }
#pragma unroll
        for (int dt = 0; dt < 8; dt++) {
            int dcol = dt * 16 + lm;
#pragma unroll
            for (int kc2 = 0; kc2 < 2; kc2++) {
                Frag16 vf;
                const unsigned short* p = Vt + (size_t)dcol * NT + kb * 64 + kc2 * 32 + lh * 8;
                vf.u[0] = *(const u32x4*)p;
                vf.u[1] = *(const u32x4*)(p + 16);
                o[dt] = wmma_bf16(pf[kc2].bf, vf.bf, o[dt]);
            }
        }
    }

    // ---- epilogue: normalize and write y[b][t][h*128+d] as bf16 ----
#pragma unroll
    for (int r = 0; r < 8; r++) {
        float inv = 1.0f / l[r];
        int trow = qbase + lh * 8 + r;
#pragma unroll
        for (int dt = 0; dt < 8; dt++) {
            int col = h * ND + dt * 16 + lm;
            y[((size_t)(b * NT + trow)) * NC + col] = f2bf(o[dt][r] * inv);
        }
    }
}

// ---------------------------------------------------------------------------
// Host launcher
// ---------------------------------------------------------------------------
extern "C" void kernel_launch(void* const* d_in, const int* in_sizes, int n_in,
                              void* d_out, int out_size, void* d_ws, size_t ws_size,
                              hipStream_t stream)
{
    const float* x      = (const float*)d_in[0];   // (4,2048,2048)
    const float* W_attn = (const float*)d_in[1];   // (2048,6144)
    const float* b_attn = (const float*)d_in[2];   // (6144,)
    const float* Aq     = (const float*)d_in[3];   // (2048,8)
    const float* Bq     = (const float*)d_in[4];   // (8,2048)
    const float* Ak     = (const float*)d_in[5];   // (2048,8)
    const float* Bk     = (const float*)d_in[6];   // (8,2048)
    const float* W_proj = (const float*)d_in[7];   // (2048,2048)
    const float* b_proj = (const float*)d_in[8];   // (2048,)
    float* out = (float*)d_out;

    // workspace carve-up
    char* p = (char*)d_ws;
    auto alloc = [&](size_t bytes) -> void* {
        void* r = (void*)p;
        p += (bytes + 255) & ~(size_t)255;
        return r;
    };
    const size_t ROWS = (size_t)NB * NT;           // 8192
    unsigned short* xb  = (unsigned short*)alloc(ROWS * NC * 2);          // 32 MB
    unsigned short* WTa = (unsigned short*)alloc((size_t)N3C * NC * 2);   // 24 MB
    unsigned short* WTp = (unsigned short*)alloc((size_t)NC * NC * 2);    //  8 MB
    unsigned short* qkv = (unsigned short*)alloc(ROWS * N3C * 2);         // 96 MB
    float*          tq  = (float*)alloc(ROWS * LORA_R * 4);
    float*          tk  = (float*)alloc(ROWS * LORA_R * 4);
    unsigned short* qh  = (unsigned short*)alloc(ROWS * NC * 2);          // 32 MB
    unsigned short* kh  = (unsigned short*)alloc(ROWS * NC * 2);          // 32 MB
    unsigned short* vh  = (unsigned short*)alloc(ROWS * NC * 2);          // 32 MB
    unsigned short* vt  = (unsigned short*)alloc(ROWS * NC * 2);          // 32 MB
    unsigned short* y   = (unsigned short*)alloc(ROWS * NC * 2);          // 32 MB

    // 1) pack activations and weights as bf16 (weights transposed)
    convert_f32_to_bf16_kernel<<<dim3((unsigned)(ROWS * NC / (256 * 8))), 256, 0, stream>>>(
        x, xb, ROWS * NC);
    transpose_f32_to_bf16<<<dim3(N3C / 32, NC / 32, 1), 256, 0, stream>>>(W_attn, WTa, NC, N3C);
    transpose_f32_to_bf16<<<dim3(NC / 32, NC / 32, 1), 256, 0, stream>>>(W_proj, WTp, NC, NC);

    // 2) qkv = x @ W_attn + b_attn   (bf16 output)
    gemm_wmma_kernel<true><<<dim3(N3C / 256, ROWS / 128), 256, 0, stream>>>(
        xb, WTa, b_attn, qkv, (int)ROWS, N3C, NC);

    // 3) LoRA rank-8 partials, then LoRA-B + RoPE + head reshape
    lora_reduce_kernel<<<dim3((unsigned)ROWS), 256, 0, stream>>>(qkv, Aq, Ak, tq, tk);
    fixup_kernel<<<dim3((unsigned)ROWS), 128, 0, stream>>>(qkv, tq, tk, Bq, Bk, qh, kh, vh);

    // 4) V -> V^T per (b,h) plane for PV B-operand
    transpose_bf16<<<dim3(ND / 32, NT / 32, NB * NH), 256, 0, stream>>>(vh, vt, NT, ND);

    // 5) flash attention
    flash_attn_kernel<<<dim3(NB * NH * (NT / 128)), 256, 0, stream>>>(qh, kh, vt, y);

    // 6) out = y @ W_proj + b_proj  (f32 output)
    gemm_wmma_kernel<false><<<dim3(NC / 256, ROWS / 128), 256, 0, stream>>>(
        y, WTp, b_proj, out, (int)ROWS, NC, NC);
}
